// Head_Motion_Predict_9955734192165
// MI455X (gfx1250) — matrix operand order repaired and checked by reading, not compiled, not run on hardware
//
#include <hip/hip_runtime.h>
#include <hip/hip_bf16.h>

typedef __attribute__((ext_vector_type(16))) _Float16 v16h;
typedef __attribute__((ext_vector_type(8)))  _Float16 v8h;
typedef __attribute__((ext_vector_type(8)))  float    v8f;
typedef __attribute__((ext_vector_type(4)))  float    v4f;

// ---------------------------------------------------------------- helpers
__device__ __forceinline__ float fast_rcp(float x) {
  return __builtin_amdgcn_rcpf(x);           // single v_rcp_f32, no Newton chain
}
__device__ __forceinline__ float fast_sigmoid(float x) {
  return fast_rcp(1.0f + __expf(-x));
}
__device__ __forceinline__ float fast_tanh(float x) {
  x = fminf(fmaxf(x, -15.0f), 15.0f);
  float e = __expf(2.0f * x);
  return (e - 1.0f) * fast_rcp(e + 1.0f);
}

// B-fragment (K x N tile, f16, 32x16) from LDS weight matrix stored [n][k], k-stride 64.
// Lanes 0-15: K {0..7,16..23} of column n=lane; lanes 16-31: K {8..15,24..31}.
__device__ __forceinline__ v16h load_bfrag(const _Float16* W, int nt, int kt, int lane) {
  const int n = nt * 16 + (lane & 15);
  const _Float16* p = W + n * 64 + kt * 32 + ((lane & 16) ? 8 : 0);
  union { v16h v; v8h h[2]; } u;
  u.h[0] = *(const v8h*)(p);
  u.h[1] = *(const v8h*)(p + 16);
  return u.v;
}

// A-fragment (16 x 32, f16) of h from LDS [m][k], k-stride 64.
// Lanes 0-15: row m=lane, K {0..7,16..23}; lanes 16-31: row m=lane-16, K {8..15,24..31}.
__device__ __forceinline__ v16h load_afrag_h(const _Float16* sH, int kt, int lane) {
  const int m  = lane & 15;
  const int k0 = kt * 32 + ((lane & 16) ? 8 : 0);
  union { v16h v; v8h h[2]; } u;
  u.h[0] = *(const v8h*)(sH + m * 64 + k0);
  u.h[1] = *(const v8h*)(sH + m * 64 + k0 + 16);
  return u.v;
}

// A-fragment of x from global f32 sequence, converted to f16.
// Fast vectorized path for the dominant IN==64 hidden-layer case (uniform branch).
__device__ __forceinline__ v16h load_afrag_x(const float* __restrict__ x, int rowBase,
                                             int sb, int st, int t, int IN, int kt, int lane) {
  const int m  = lane & 15;
  const int hi = (lane >> 4) & 1;
  const float* base = x + (size_t)(rowBase + m) * sb + (size_t)t * st;
  const int k0 = kt * 32 + hi * 8;
  v16h a;
  if (IN == 64) {                      // uniform: EXEC stays all-ones
    v4f x0 = *(const v4f*)(base + k0);
    v4f x1 = *(const v4f*)(base + k0 + 4);
    v4f x2 = *(const v4f*)(base + k0 + 16);
    v4f x3 = *(const v4f*)(base + k0 + 20);
#pragma unroll
    for (int i = 0; i < 4; ++i) {
      a[i]      = (_Float16)x0[i];
      a[i + 4]  = (_Float16)x1[i];
      a[i + 8]  = (_Float16)x2[i];
      a[i + 12] = (_Float16)x3[i];
    }
  } else {                             // zero-padded path (IN = 1 or 35)
#pragma unroll
    for (int i = 0; i < 8; ++i) {
      int ka = k0 + i, kb = k0 + 16 + i;
      a[i]     = (_Float16)((ka < IN) ? base[ka] : 0.0f);
      a[i + 8] = (_Float16)((kb < IN) ? base[kb] : 0.0f);
    }
  }
  return a;
}

#define WMMA(a, b, c) __builtin_amdgcn_wmma_f32_16x16x32_f16(false, (a), false, (b), (short)0, (c), false, false)

// ---------------------------------------------------------------- GRU layer
// One launch = one GRU layer, internal loop over T timesteps (recurrence).
// 64 blocks x 128 threads (4 waves). Block owns 16 batch rows; wave w owns gate
// column tiles {w, w+4, w+8} (r/z/n triple) -> h columns 16w..16w+15 stay in regs.
__global__ __launch_bounds__(128) void gru_layer_kernel(
    const float* __restrict__ xin, int T, int IN, int sb, int st, int nKI,
    const float* __restrict__ Wih, const float* __restrict__ Whh,
    const float* __restrict__ bih, const float* __restrict__ bhh,
    float* __restrict__ out)
{
  __shared__ __align__(16) _Float16 sWih[192 * 64];
  __shared__ __align__(16) _Float16 sWhh[192 * 64];
  __shared__ __align__(16) _Float16 sH[16 * 64];

  const int tid     = threadIdx.x;
  const int wave    = tid >> 5;
  const int lane    = tid & 31;
  const int rowBase = blockIdx.x << 4;

  // stage weights f32 -> f16 in LDS, Wih zero-padded in K up to 64
  for (int i = tid; i < 192 * 64; i += 128) {
    int n = i >> 6, k = i & 63;
    sWih[i] = (_Float16)((k < IN) ? Wih[n * IN + k] : 0.0f);
    sWhh[i] = (_Float16)Whh[i];
  }
  for (int i = tid; i < 16 * 64; i += 128) sH[i] = (_Float16)0.0f;
  __syncthreads();

  // hoist weight fragments to registers (constant over t)
  v16h fih[3][2], fhh[3][2];
#pragma unroll
  for (int g = 0; g < 3; ++g) {
#pragma unroll
    for (int kt = 0; kt < 2; ++kt) {
      fih[g][kt] = load_bfrag(sWih, wave + 4 * g, kt, lane);
      fhh[g][kt] = load_bfrag(sWhh, wave + 4 * g, kt, lane);
    }
  }
  const int colg = (wave << 4) + (lane & 15);
  const float biR = bih[colg],       bhR = bhh[colg];
  const float biZ = bih[64 + colg],  bhZ = bhh[64 + colg];
  const float biN = bih[128 + colg], bhN = bhh[128 + colg];

  v8f hprev = {};
  for (int t = 0; t < T; ++t) {
    // gi = xt @ Wih^T
    v8f aR = {}, aZ = {}, aN = {};
    for (int kt = 0; kt < nKI; ++kt) {
      v16h a = load_afrag_x(xin, rowBase, sb, st, t, IN, kt, lane);
      aR = WMMA(a, fih[0][kt], aR);
      aZ = WMMA(a, fih[1][kt], aZ);
      aN = WMMA(a, fih[2][kt], aN);
    }
    // prefetch next timestep's x row (global_prefetch_b8)
    if (t + 1 < T) {
      const float* nxt = xin + (size_t)(rowBase + (lane & 15)) * sb + (size_t)(t + 1) * st;
      __builtin_prefetch(nxt, 0, 1);
    }
    // gh = h @ Whh^T (h from shared tile written last step)
    v8f hR = {}, hZ = {}, hN = {};
#pragma unroll
    for (int kt = 0; kt < 2; ++kt) {
      v16h a = load_afrag_h(sH, kt, lane);
      hR = WMMA(a, fhh[0][kt], hR);
      hZ = WMMA(a, fhh[1][kt], hZ);
      hN = WMMA(a, fhh[2][kt], hN);
    }
    __syncthreads();   // all waves done reading sH before it is overwritten

    v8f hnew;
#pragma unroll
    for (int e = 0; e < 8; ++e) {
      float r = fast_sigmoid(aR[e] + biR + hR[e] + bhR);
      float z = fast_sigmoid(aZ[e] + biZ + hZ[e] + bhZ);
      float n = fast_tanh(aN[e] + biN + r * (hN[e] + bhN));
      hnew[e] = (1.0f - z) * n + z * hprev[e];
    }
    hprev = hnew;

    const int mBase = (lane >> 4) << 3;   // C layout: VGPR e -> row e (+8 for hi lanes)
#pragma unroll
    for (int e = 0; e < 8; ++e) {
      int m = mBase + e;
      out[(size_t)(rowBase + m) * T * 64 + (size_t)t * 64 + colg] = hnew[e];
      sH[m * 64 + colg] = (_Float16)hnew[e];
    }
    __syncthreads();   // sH fully written before next step's gh GEMM
  }
}

// ---------------------------------------------------------------- front end
// conv1d (k=10, VALID: 19->10) + training-mode BN over (batch, pos), per stream.
__global__ __launch_bounds__(1024) void frontend_kernel(
    const float* __restrict__ x,
    const float* w0, const float* w1, const float* w2,
    const float* b0, const float* b1, const float* b2,
    const float* g0, const float* g1, const float* g2,
    const float* e0, const float* e1, const float* e2,
    float* __restrict__ s)
{
  __shared__ float red1[1024];
  __shared__ float red2[1024];
  const int c = blockIdx.x;      // 0=theta 1=phi 2=psi
  const int b = threadIdx.x;
  const float* w   = (c == 0) ? w0 : (c == 1) ? w1 : w2;
  const float bias = ((c == 0) ? b0 : (c == 1) ? b1 : b2)[0];
  const float gg   = ((c == 0) ? g0 : (c == 1) ? g1 : g2)[0];
  const float bb   = ((c == 0) ? e0 : (c == 1) ? e1 : e2)[0];

  float xi[19], wv[10], yv[10];
#pragma unroll
  for (int i = 0; i < 19; ++i) xi[i] = x[(size_t)b * 57 + (size_t)c * 19 + i];
#pragma unroll
  for (int i = 0; i < 10; ++i) wv[i] = w[i];
  float s1 = 0.f, s2 = 0.f;
#pragma unroll
  for (int tp = 0; tp < 10; ++tp) {
    float acc = bias;
#pragma unroll
    for (int k = 0; k < 10; ++k) acc += xi[tp + k] * wv[k];
    yv[tp] = acc; s1 += acc; s2 += acc * acc;
  }
  red1[b] = s1; red2[b] = s2;
  __syncthreads();
  for (int off = 512; off > 0; off >>= 1) {
    if (b < off) { red1[b] += red1[b + off]; red2[b] += red2[b + off]; }
    __syncthreads();
  }
  const float inv  = 1.0f / 10240.0f;
  const float mean = red1[0] * inv;
  const float var  = red2[0] * inv - mean * mean;
  const float rstd = rsqrtf(var + 1e-5f);
#pragma unroll
  for (int tp = 0; tp < 10; ++tp)
    s[(size_t)b * 30 + c * 10 + tp] = gg * (yv[tp] - mean) * rstd + bb;
}

// ---------------------------------------------------------------- cnn1: real 2D conv
// out[b,o,j] = sum_{u,v<30} r[b,u,j+v] * w[o,u,v] + bias[o];  r: (B,30,64), out: (B,30,35)
__global__ __launch_bounds__(256) void conv1_kernel(
    const float* __restrict__ r, const float* __restrict__ w,
    const float* __restrict__ bias, float* __restrict__ y)
{
  __shared__ float sR[30 * 64];
  const int b = blockIdx.x;
  for (int i = threadIdx.x; i < 1920; i += 256) sR[i] = r[(size_t)b * 1920 + i];
  __syncthreads();
  for (int idx = threadIdx.x; idx < 1050; idx += 256) {
    const int o = idx / 35, j = idx % 35;
    const float* wo = w + o * 900;
    float acc = bias[o];
    for (int u = 0; u < 30; ++u) {
      const float* rr = sR + u * 64 + j;
      const float* ww = wo + u * 30;
#pragma unroll
      for (int v = 0; v < 30; ++v) acc += rr[v] * ww[v];
    }
    y[(size_t)b * 1050 + idx] = acc;
  }
}

// cnn2-6: input rows are broadcast copies -> collapse kernel over u: W2[o,v] = sum_u w[o,u,v]
__global__ void reduce_w_kernel(const float* __restrict__ w, float* __restrict__ W2) {
  const int t = threadIdx.x;
  if (t < 900) {
    const int o = t / 30, v = t % 30;
    float acc = 0.f;
    for (int u = 0; u < 30; ++u) acc += w[o * 900 + u * 30 + v];
    W2[t] = acc;
  }
}

__global__ __launch_bounds__(256) void conv_bcast_kernel(
    const float* __restrict__ r, const float* __restrict__ W2,
    const float* __restrict__ bias, float* __restrict__ y)
{
  __shared__ float sR[64];
  __shared__ float sW[900];
  const int b = blockIdx.x;
  if (threadIdx.x < 64) sR[threadIdx.x] = r[(size_t)b * 64 + threadIdx.x];
  for (int i = threadIdx.x; i < 900; i += 256) sW[i] = W2[i];
  __syncthreads();
  for (int idx = threadIdx.x; idx < 1050; idx += 256) {
    const int o = idx / 35, j = idx % 35;
    float acc = bias[o];
#pragma unroll
    for (int v = 0; v < 30; ++v) acc += sW[o * 30 + v] * sR[j + v];
    y[(size_t)b * 1050 + idx] = acc;
  }
}

// ---------------------------------------------------------------- batch-norm (training stats)
__global__ __launch_bounds__(256) void bn_stats_kernel(const float* __restrict__ y,
                                                       float* __restrict__ stats) {
  __shared__ float r1[256], r2[256];
  const int c = blockIdx.x;
  float s1 = 0.f, s2 = 0.f;
  for (int i = threadIdx.x; i < 35840; i += 256) {
    int b = i / 35, j = i % 35;
    float v = y[(size_t)b * 1050 + c * 35 + j];
    s1 += v; s2 += v * v;
  }
  r1[threadIdx.x] = s1; r2[threadIdx.x] = s2;
  __syncthreads();
  for (int off = 128; off > 0; off >>= 1) {
    if (threadIdx.x < off) { r1[threadIdx.x] += r1[threadIdx.x + off];
                             r2[threadIdx.x] += r2[threadIdx.x + off]; }
    __syncthreads();
  }
  if (threadIdx.x == 0) {
    const float inv  = 1.0f / 35840.0f;
    const float mean = r1[0] * inv;
    const float var  = r2[0] * inv - mean * mean;
    stats[c]      = mean;
    stats[32 + c] = rsqrtf(var + 1e-5f);
  }
}

__global__ __launch_bounds__(256) void bn_apply_kernel(
    float* __restrict__ y, const float* __restrict__ stats,
    const float* __restrict__ g, const float* __restrict__ beta,
    float* __restrict__ feat)
{
  const int e = blockIdx.x * 256 + threadIdx.x;
  if (e >= 1024 * 1050) return;
  const int b   = e / 1050;
  const int rem = e - b * 1050;
  const int c   = rem / 35;
  const int j   = rem - c * 35;
  const float v = g[c] * (y[e] - stats[c]) * stats[32 + c] + beta[c];
  y[e] = v;
  if (c == 0) feat[b * 35 + j] = v;   // feat = y[:,0,:,:]
}

// ---------------------------------------------------------------- max-pool + FNN + relu
__global__ __launch_bounds__(64) void pool_fnn_kernel(
    const float* __restrict__ y, const float* __restrict__ fw,
    const float* __restrict__ fb, float* __restrict__ outp)
{
  __shared__ float sPool[30];
  const int b = blockIdx.x;
  const int t = threadIdx.x;
  if (t < 30) {
    float m = -3.4e38f;
    const float* yp = y + (size_t)b * 1050 + t * 35;
#pragma unroll
    for (int j = 0; j < 35; ++j) m = fmaxf(m, yp[j]);
    sPool[t] = m;
  }
  __syncthreads();
  if (t < 27) {
    float acc = fb[t];
#pragma unroll
    for (int o = 0; o < 30; ++o) acc += sPool[o] * fw[t * 30 + o];
    outp[(size_t)b * 27 + t] = fmaxf(acc, 0.0f);
  }
}

// ---------------------------------------------------------------- host orchestration
extern "C" void kernel_launch(void* const* d_in, const int* in_sizes, int n_in,
                              void* d_out, int out_size, void* d_ws, size_t ws_size,
                              hipStream_t stream) {
  (void)in_sizes; (void)out_size; (void)ws_size;
  // Flattened input order (alphabetical dict flatten):
  // 0-5 bn_{phi,psi,theta}_{b,g}; 6-29 cnn1..6 {b,beta,g,w}; 30-35 conv_{phi,psi,theta}_{b,w};
  // 36 fnn_b; 37 fnn_w; 38-67 gru1..6 {Whh,Wih,Wih0,bhh,bih}; 68 x
  const float* P[69] = {};
  for (int i = 0; i < 69 && i < n_in; ++i) P[i] = (const float*)d_in[i];
  const float* x = P[68];

  float* ws   = (float*)d_ws;
  float* s    = ws;                  // 1024*30
  float* seqA = s + 30720;           // 1024*30*64
  float* seqB = seqA + 1966080;      // 1024*30*64
  float* y    = seqB + 1966080;      // 1024*30*35
  float* feat = y + 1075200;         // 1024*35
  float* W2   = feat + 35840;        // 900
  float* stats= W2 + 900;            // 64

  frontend_kernel<<<3, 1024, 0, stream>>>(x,
      P[35], P[31], P[33],           // conv w: theta, phi, psi
      P[34], P[30], P[32],           // conv b
      P[5],  P[1],  P[3],            // bn gamma
      P[4],  P[0],  P[2],            // bn beta
      s);

  auto gru_base = [&](int k) { return 38 + (k - 1) * 5; };
  auto cnn_base = [&](int k) { return 6 + (k - 1) * 4; };

  auto run_gru = [&](const float* in0, int T, int IN0, int sb0, int st0, int k) -> float* {
    const int gb = gru_base(k);
    const float* Whh  = P[gb + 0];
    const float* Wih  = P[gb + 1];
    const float* Wih0 = P[gb + 2];
    const float* bhh  = P[gb + 3];
    const float* bih  = P[gb + 4];
    int nKI0 = (IN0 + 31) / 32; if (nKI0 > 2) nKI0 = 2;
    gru_layer_kernel<<<64, 128, 0, stream>>>(in0, T, IN0, sb0, st0, nKI0,
                                             Wih0, Whh, bih, bhh, seqA);
    float* cur = seqA; float* nxt = seqB;
    for (int l = 1; l < 30; ++l) {
      gru_layer_kernel<<<64, 128, 0, stream>>>(cur, T, 64, T * 64, 64, 2,
          Wih + (size_t)(l - 1) * 192 * 64,
          Whh + (size_t)l * 192 * 64,
          bih + (size_t)l * 192,
          bhh + (size_t)l * 192,
          nxt);
      float* tmp = cur; cur = nxt; nxt = tmp;
    }
    return cur;
  };

  // block 1: gru over seq len 30 (input dim 1), full 2D conv, BN
  float* r = run_gru(s, 30, 1, 30, 1, 1);
  {
    const int cb = cnn_base(1);
    conv1_kernel<<<1024, 256, 0, stream>>>(r, P[cb + 3], P[cb + 0], y);
    bn_stats_kernel<<<30, 256, 0, stream>>>(y, stats);
    bn_apply_kernel<<<4200, 256, 0, stream>>>(y, stats, P[cb + 2], P[cb + 1], feat);
  }
  // blocks 2..6: gru over seq len 1 (input dim 35), broadcast conv, BN
  for (int k = 2; k <= 6; ++k) {
    r = run_gru(feat, 1, 35, 35, 35, k);
    const int cb = cnn_base(k);
    reduce_w_kernel<<<1, 1024, 0, stream>>>(P[cb + 3], W2);
    conv_bcast_kernel<<<1024, 256, 0, stream>>>(r, W2, P[cb + 0], y);
    bn_stats_kernel<<<30, 256, 0, stream>>>(y, stats);
    bn_apply_kernel<<<4200, 256, 0, stream>>>(y, stats, P[cb + 2], P[cb + 1], feat);
  }
  pool_fnn_kernel<<<1024, 64, 0, stream>>>(y, P[37], P[36], (float*)d_out);
}
